// Reconstructor_43757126812349
// MI455X (gfx1250) — matrix operand order repaired and checked by reading, not compiled, not run on hardware
//
#include <hip/hip_runtime.h>
#include <hip/hip_bf16.h>

typedef __attribute__((ext_vector_type(16))) __bf16 v16bf;
typedef __attribute__((ext_vector_type(8)))  __bf16 v8bf;
typedef __attribute__((ext_vector_type(8)))  float  v8f;

#define N_CAT     32
#define DIM_NUM   64
#define DIM_TOKEN 512
#define BATCH     4096
#define ROW_STRIDE (96 * 512)   // (DIM_NUM + N_CAT) * DIM_TOKEN
#define M_SUB     4             // 4 x 16 = 64 batch rows per workgroup block
#define LDS_ROWB  80            // padded LDS row stride (bytes): 20 dwords,
                                // 16B aligned, gcd(20,64)=4 -> rows 0..15 hit
                                // 16 distinct bank groups (conflict-free)
#define LDS_MAT   (64 * LDS_ROWB)        // one 64x32 bf16 matrix: 5120 B
#define LDS_BUF   (2 * LDS_MAT)          // hi + lo

// ---------------------------------------------------------------------------
// Kernel 1: re_x_num[b,n] = dot(h[b, n*512 : (n+1)*512], weight[n,:])
// Pure bandwidth: one wave32 per (b,n), float4 loads, shuffle reduction.
// ---------------------------------------------------------------------------
__global__ __launch_bounds__(256) void renum_kernel(
    const float* __restrict__ h, const float* __restrict__ weight,
    float* __restrict__ out) {
  const int wavesPerBlock = blockDim.x >> 5;
  const int wid  = blockIdx.x * wavesPerBlock + (threadIdx.x >> 5);
  const int lane = threadIdx.x & 31;
  const int b = wid >> 6;          // / 64
  const int n = wid & 63;
  if (b >= BATCH) return;

  const float* hp = h + (size_t)b * ROW_STRIDE + n * DIM_TOKEN;
  const float* wp = weight + n * DIM_TOKEN;

  float acc = 0.0f;
#pragma unroll
  for (int k = 0; k < DIM_TOKEN; k += 128) {
    const float4 hv = *reinterpret_cast<const float4*>(hp + k + lane * 4);
    const float4 wv = *reinterpret_cast<const float4*>(wp + k + lane * 4);
    acc = fmaf(hv.x, wv.x, acc);
    acc = fmaf(hv.y, wv.y, acc);
    acc = fmaf(hv.z, wv.z, acc);
    acc = fmaf(hv.w, wv.w, acc);
  }
#pragma unroll
  for (int off = 16; off > 0; off >>= 1) acc += __shfl_xor(acc, off, 32);
  if (lane == 0) out[wid] = acc;   // wid == b*64 + n : exact output layout
}

// ---------------------------------------------------------------------------
// Kernel 2: 32 grouped GEMMs  C_i = H_i (4096x512) * W_i^T (512xd_i) + b_i
// Split-bf16 WMMA: x = hi + lo (bf16 each); C += Ahi*Bhi + Ahi*Blo + Alo*Bhi.
// Workgroup = 8 waves = 64 rows x 128 cols. The A panel (64x32 per K-step) is
// split to bf16 hi/lo cooperatively ONCE into LDS (double-buffered) and read
// back via ds_load_b128 in WMMA A-layout order by all 8 waves.
// ---------------------------------------------------------------------------
struct CatParams {
  const float* W[N_CAT];
  const float* bias[N_CAT];
  long long    outOff[N_CAT];     // element offset into d_out
  int          d[N_CAT];
  int          wgPrefix[N_CAT + 1];
  int          nBlocks[N_CAT];    // ceil(ceil(d/16) / 8)
};

// Load 8 contiguous fp32, split each into bf16 hi/lo at vector slots BASE..BASE+7
template <int BASE>
__device__ __forceinline__ void split8(const float* __restrict__ p,
                                       v16bf& hi, v16bf& lo) {
  const float4 f0 = *reinterpret_cast<const float4*>(p);
  const float4 f1 = *reinterpret_cast<const float4*>(p + 4);
  const float v[8] = {f0.x, f0.y, f0.z, f0.w, f1.x, f1.y, f1.z, f1.w};
#pragma unroll
  for (int j = 0; j < 8; ++j) {
    const __bf16 hb = (__bf16)v[j];
    hi[BASE + j] = hb;
    lo[BASE + j] = (__bf16)(v[j] - (float)hb);
  }
}

__global__ __launch_bounds__(256) void recat_wmma_kernel(
    const float* __restrict__ h, float* __restrict__ out, CatParams P) {
  __shared__ __align__(16) unsigned char smem[2 * LDS_BUF];   // 20480 B

  // ---- locate category / block (grid is exact: no early exit, barriers ok) --
  const int wg = blockIdx.x;
  int cat = 0;
  while (cat < N_CAT - 1 && wg >= P.wgPrefix[cat + 1]) ++cat;
  const int d       = P.d[cat];
  const int nT      = (d + 15) >> 4;
  const int nBlocks = P.nBlocks[cat];
  const int local   = wg - P.wgPrefix[cat];
  const int mBlk    = local / nBlocks;      // consecutive wgs share A rows
  const int nBlk    = local % nBlocks;
  const int mBase   = mBlk * (16 * M_SUB);

  const int wv = threadIdx.x >> 5;
  int nTile = nBlk * 8 + wv;
  if (nTile >= nT) nTile = nT - 1;          // spare waves duplicate (benign)
  const int nBase = nTile * 16;

  const int lane = threadIdx.x & 31;
  const int g    = lane >> 4;               // half-wave group (0/1)
  const int r    = lane & 15;

  const size_t tokOff = (size_t)(DIM_NUM + cat) * DIM_TOKEN;

  // Cooperative A mapping: thread t -> row t/4 of the 64-row panel,
  // k-chunk (t%4)*8 of the 32-wide K step (8 fp32 = 32B contiguous).
  const int cRow = threadIdx.x >> 2;
  const int cK   = (threadIdx.x & 3) * 8;
  const float* aP = h + (size_t)(mBase + cRow) * ROW_STRIDE + tokOff + cK;

  // B: column n of W^T = row n of W; B layout: lane=N, elem e<8 -> K = 8g+e, e>=8 -> K = 16+8g+(e-8).
  // Clamp OOB rows (partial last N tile) so EXEC stays uniform; mask on store.
  const int wrow = (nBase + r < d) ? (nBase + r) : (d - 1);
  const float* bRow = P.W[cat] + (size_t)wrow * DIM_TOKEN;
  const int bK0 = g * 8;

  v8f acc[M_SUB] = {};
  for (int kt = 0; kt < DIM_TOKEN / 32; ++kt) {
    const int kb = kt * 32;
    unsigned char* buf = smem + (kt & 1) * LDS_BUF;   // double buffer

    // ---- cooperative split of the 64x32 A panel into LDS (hi @0, lo @MAT) --
    {
      const float4 f0 = *reinterpret_cast<const float4*>(aP + kb);
      const float4 f1 = *reinterpret_cast<const float4*>(aP + kb + 4);
      const float v[8] = {f0.x, f0.y, f0.z, f0.w, f1.x, f1.y, f1.z, f1.w};
      v8bf hv, lv;
#pragma unroll
      for (int j = 0; j < 8; ++j) {
        const __bf16 hb = (__bf16)v[j];
        hv[j] = hb;
        lv[j] = (__bf16)(v[j] - (float)hb);
      }
      *reinterpret_cast<v8bf*>(buf + cRow * LDS_ROWB + cK * 2)           = hv;
      *reinterpret_cast<v8bf*>(buf + LDS_MAT + cRow * LDS_ROWB + cK * 2) = lv;
    }

    // ---- per-wave B split (private: each wave owns different W rows) ------
    v16bf bhi, blo;
    split8<0>(bRow + kb + bK0,     bhi, blo);
    split8<8>(bRow + kb + bK0 + 16, bhi, blo);

    __syncthreads();   // one barrier per K-step (double buffer covers WAR)

    // ---- 4 sub-tiles: A from LDS in WMMA layout, 3-pass split-bf16 MMA ----
#pragma unroll
    for (int s = 0; s < M_SUB; ++s) {
      // A layout: lane r = M row, elems 0..7 at K byte 16g, 8..15 at +32
      const unsigned char* ah = buf + (16 * s + r) * LDS_ROWB + g * 16;
      const unsigned char* al = ah + LDS_MAT;
      v16bf ahi, alo;
      *reinterpret_cast<float4*>(&ahi) =
          *reinterpret_cast<const float4*>(ah);
      *(reinterpret_cast<float4*>(&ahi) + 1) =
          *reinterpret_cast<const float4*>(ah + 32);
      *reinterpret_cast<float4*>(&alo) =
          *reinterpret_cast<const float4*>(al);
      *(reinterpret_cast<float4*>(&alo) + 1) =
          *reinterpret_cast<const float4*>(al + 32);

      acc[s] = __builtin_amdgcn_wmma_f32_16x16x32_bf16(
          false, ahi, false, bhi, (short)0, acc[s], false, false);
      acc[s] = __builtin_amdgcn_wmma_f32_16x16x32_bf16(
          false, ahi, false, blo, (short)0, acc[s], false, false);
      acc[s] = __builtin_amdgcn_wmma_f32_16x16x32_bf16(
          false, alo, false, bhi, (short)0, acc[s], false, false);
    }
  }

  // C/D layout: VGPR v, lanes 0-15 -> M = v, lanes 16-31 -> M = v + 8; N = r.
  const int col = nBase + r;
  if (col < d) {
    const float bv = P.bias[cat][col];
#pragma unroll
    for (int s = 0; s < M_SUB; ++s) {
      float* o = out + P.outOff[cat]
                     + (size_t)(mBase + 16 * s + g * 8) * d + col;
#pragma unroll
      for (int v = 0; v < 8; ++v) o[(size_t)v * d] = acc[s][v] + bv;
    }
  }
}

// ---------------------------------------------------------------------------
extern "C" void kernel_launch(void* const* d_in, const int* in_sizes, int n_in,
                              void* d_out, int out_size, void* d_ws,
                              size_t ws_size, hipStream_t stream) {
  (void)in_sizes; (void)n_in; (void)out_size; (void)d_ws; (void)ws_size;
  static const int CATS[N_CAT] = {2, 3, 5, 8, 10, 12, 16, 20, 24, 32, 40, 48,
                                  64, 80, 96, 100, 128, 150, 160, 200, 256,
                                  300, 320, 400, 512, 600, 640, 700, 768, 800,
                                  900, 1000};
  const float* h      = (const float*)d_in[0];
  const float* weight = (const float*)d_in[1];
  float*       out    = (float*)d_out;

  CatParams P;
  long long off = (long long)BATCH * DIM_NUM;   // re_x_num comes first
  int wp = 0;
  P.wgPrefix[0] = 0;
  for (int i = 0; i < N_CAT; ++i) {
    P.W[i]      = (const float*)d_in[2 + i];
    P.bias[i]   = (const float*)d_in[2 + N_CAT + i];
    P.outOff[i] = off;
    P.d[i]      = CATS[i];
    off += (long long)BATCH * CATS[i];
    const int nT = (CATS[i] + 15) / 16;
    P.nBlocks[i] = (nT + 7) / 8;
    wp += (BATCH / (16 * M_SUB)) * P.nBlocks[i];
    P.wgPrefix[i + 1] = wp;
  }

  // Part 1: 4096*64 waves, 8 waves / 256-thread block
  renum_kernel<<<(BATCH * DIM_NUM) / 8, 256, 0, stream>>>(h, weight, out);

  // Part 2: one workgroup (8 waves) per 64x128 block
  recat_wmma_kernel<<<wp, 256, 0, stream>>>(h, out, P);
}
